// mLSTM_11965778886689
// MI455X (gfx1250) — matrix operand order repaired and checked by reading, not compiled, not run on hardware
//
#include <hip/hip_runtime.h>
#include <math.h>
#include <stdint.h>

typedef __attribute__((ext_vector_type(16))) _Float16 v16h;
typedef __attribute__((ext_vector_type(8)))  _Float16 v8h;
typedef __attribute__((ext_vector_type(8)))  float    v8f;

#define BX 8
#define TX 512
#define DX 128
#define HX 128
#define CS 32                         // chunk size (timesteps per chunk)
#define NCH (TX/CS)                   // 16 chunks
#define MTOT (BX*TX)                  // 4096 rows in the projection GEMM
#define GSTRIDE ((size_t)MTOT*HX)     // per-projection gate plane (f32 elems)

// ---------------------------------------------------------------------------
// WMMA fragment helpers (16-bit A 16x32, contiguous k = 16*hs + s per lane half; row-major source rows)
// ---------------------------------------------------------------------------
__device__ inline v16h load_afrag(const _Float16* row, int k0, int hs) {
  v8h a0 = *(const v8h*)(row + k0 + hs * 16);
  v8h a1 = *(const v8h*)(row + k0 + hs * 16 + 8);
  v16h a;
#pragma unroll
  for (int h = 0; h < 8; ++h) { a[h] = a0[h]; a[h + 8] = a1[h]; }
  return a;
}

__device__ inline v8f wmma_f16(v16h a, v16h b, v8f c) {
  return __builtin_amdgcn_wmma_f32_16x16x32_f16(false, a, false, b,
                                                (short)0, c, false, false);
}

// ---------------------------------------------------------------------------
__global__ void cvt_f32_f16(const float* __restrict__ src,
                            _Float16* __restrict__ dst, int n) {
  int i = blockIdx.x * blockDim.x + threadIdx.x;
  if (i < n) dst[i] = (_Float16)src[i];
}

__global__ void pack_w16(const float* __restrict__ Wq, const float* __restrict__ Wk,
                         const float* __restrict__ Wv, const float* __restrict__ Wi,
                         const float* __restrict__ Wf, const float* __restrict__ Wo,
                         _Float16* __restrict__ w16, int layer) {
  int i = blockIdx.x * blockDim.x + threadIdx.x;
  const int per = HX * DX;
  if (i >= 6 * per) return;
  int p = i / per, off = i % per;
  const float* W = (p == 0) ? Wq : (p == 1) ? Wk : (p == 2) ? Wv
                 : (p == 3) ? Wi : (p == 4) ? Wf : Wo;
  w16[i] = (_Float16)W[(size_t)layer * per + off];
}

// ---------------------------------------------------------------------------
// Projection GEMM: g[p] = act_p( A(4096x128,f16) * W_p^T(128x128,f16) + b_p )
// ---------------------------------------------------------------------------
__global__ __launch_bounds__(256)
void proj_gemm(const _Float16* __restrict__ A,
               const _Float16* __restrict__ W16,
               const float* __restrict__ bq, const float* __restrict__ bk,
               const float* __restrict__ bv, const float* __restrict__ bi,
               const float* __restrict__ bf, const float* __restrict__ bo,
               float* __restrict__ g, int layer) {
  const int proj  = blockIdx.y;
  const int tileM = blockIdx.x;
  const int wave  = threadIdx.x >> 5;
  const int lane  = threadIdx.x & 31;
  const int hs    = lane >> 4;
  const int l16   = lane & 15;

  const int row = tileM * 16 + l16;
  const int col = wave  * 16 + l16;

  const _Float16* arow = A + (size_t)row * DX;
  const _Float16* wrow = W16 + (size_t)proj * HX * DX + (size_t)col * DX;

  v8f c = {};
#pragma unroll
  for (int k0 = 0; k0 < DX; k0 += 32) {
    v16h a = load_afrag(arow, k0, hs);
    v16h b = *(const v16h*)(wrow + k0 + hs * 16);
    c = wmma_f16(a, b, c);
  }

  const float* bias = (proj == 0) ? bq : (proj == 1) ? bk : (proj == 2) ? bv
                    : (proj == 3) ? bi : (proj == 4) ? bf : bo;
  const float bb = bias[(size_t)layer * HX + col];
  float* gout = g + (size_t)proj * GSTRIDE;
#pragma unroll
  for (int r = 0; r < 8; ++r) {
    int m = tileM * 16 + r + hs * 8;
    float val = c[r] + bb;
    if (proj == 1)      val *= 0.08838834764831845f;          // k * 1/sqrt(H)
    else if (proj == 3) val = __expf(val);                    // i = exp(.)
    else if (proj >= 4) val = 1.0f / (1.0f + __expf(-val));   // f,o = sigmoid
    gout[(size_t)m * HX + col] = val;
  }
}

// ---------------------------------------------------------------------------
// Chunkwise-parallel mLSTM scan. One block per batch, 256 threads = 8 waves.
// Next chunk's gates are async-copied global->LDS (ASYNCcnt) by waves 5..7
// during the GEMM phases, hiding the gate fetch behind WMMA work.
// ---------------------------------------------------------------------------
__global__ __launch_bounds__(256)
void mlstm_chunk(const float* __restrict__ g, float* __restrict__ out) {
  extern __shared__ char smem[];
  float*    Cs  = (float*)   (smem);                 // 128*128 f32  65536
  _Float16* Qh  = (_Float16*)(smem + 65536);         // 32*128 f16    8192
  _Float16* Kh  = (_Float16*)(smem + 73728);         // 32*128 f16    8192
  _Float16* Kt  = (_Float16*)(smem + 81920);         // 128*32 f16    8192
  _Float16* Vt  = (_Float16*)(smem + 90112);         // 128*32 f16    8192
  float*    Sc  = (float*)   (smem + 98304);         // 32*32  f32    4096
  float*    Uf  = (float*)   (smem + 102400);        // 32*128 f32   16384
  float*    Gs  = (float*)   (smem + 118784);        // 32*128 f32   16384
  float*    Nr  = (float*)   (smem + 135168);        // 32*128 f32   16384
  float*    Hi  = (float*)   (smem + 151552);        // 32*128 f32   16384
  float*    qf  = (float*)   (smem + 167936);        // 32*128 f32   16384
  float*    np  = (float*)   (smem + 184320);        // 128    f32     512
  float*    Ge  = (float*)   (smem + 184832);        // 128    f32     512
  float*    den = (float*)   (smem + 185344);        // 32     f32     128
  float*    stg = (float*)   (smem + 185472);        // 5*32*128 f32 81920

  const int b    = blockIdx.x;
  const int tid  = threadIdx.x;
  const int wave = tid >> 5;
  const int lane = tid & 31;
  const int hs   = lane >> 4;
  const int l16  = lane & 15;

  for (int e = tid; e < HX * HX; e += 256) Cs[e] = 0.0f;
  if (tid < HX) np[tid] = 0.0f;
  __syncthreads();

  for (int chn = 0; chn < NCH; ++chn) {
    const int t0 = chn * CS;

    // ---------------- Phase A: per-channel gate prep (threads 0..127) ----
    if (tid < HX) {
      const int i = tid;
      float grun = 1.0f, nrun = 0.0f;
      const size_t gb = (size_t)(b * TX + t0) * HX + i;
      for (int s = 0; s < CS; ++s) {
        float qv, kv, vv, iv, fv;
        if (chn == 0) {                      // first chunk: direct global reads
          const size_t off = gb + (size_t)s * HX;
          qv = g[0 * GSTRIDE + off];
          kv = g[1 * GSTRIDE + off];
          vv = g[2 * GSTRIDE + off];
          iv = g[3 * GSTRIDE + off];
          fv = g[4 * GSTRIDE + off];
        } else {                             // staged by async copy last chunk
          const int off = s * HX + i;
          qv = stg[0 * CS * HX + off];
          kv = stg[1 * CS * HX + off];
          vv = stg[2 * CS * HX + off];
          iv = stg[3 * CS * HX + off];
          fv = stg[4 * CS * HX + off];
        }
        grun *= fv;
        grun = fmaxf(grun, 1e-30f);          // underflow guard
        const float ginv = 1.0f / grun;
        Gs[s * HX + i] = grun;
        Uf[s * HX + i] = iv * vv * ginv;
        nrun = fmaf(iv * kv, ginv, nrun);
        Nr[s * HX + i] = nrun;
        Qh[s * HX + i] = (_Float16)qv;
        qf[s * HX + i] = qv;
        Kh[s * HX + i] = (_Float16)kv;
        Kt[i * CS + s] = (_Float16)kv;
      }
      Ge[i] = grun;
      for (int u = 0; u < CS; ++u)
        Vt[i * CS + u] = (_Float16)(grun * Uf[u * HX + i]);  // ratio<=1: f16-safe
    }
    __syncthreads();

    // Prefetch current chunk's o-gate slab (consumed in phase D).
    __builtin_prefetch((const char*)(g + 5 * GSTRIDE +
                                     (size_t)(b * TX + t0) * HX) +
                           (size_t)tid * 64, 0, 0);

    // ---------------- Phase B/C: score GEMM + inter GEMM + den + async ---
    if (wave < 4) {
      // Sc(32x32) = Q K^T  (one 16x16 tile per wave)
      const int tS = wave >> 1, tU = wave & 1;
      v8f c = {};
#pragma unroll
      for (int k0 = 0; k0 < HX; k0 += 32) {
        v16h a = load_afrag(Qh + (tS * 16 + l16) * HX, k0, hs);
        v16h bb = *(const v16h*)(Kh + (tU * 16 + l16) * HX + k0 + hs * 16);
        c = wmma_f16(a, bb, c);
      }
#pragma unroll
      for (int r = 0; r < 8; ++r) {
        int s_ = tS * 16 + r + hs * 8;
        int u_ = tU * 16 + l16;
        Sc[s_ * CS + u_] = (u_ <= s_) ? c[r] : 0.0f;   // causal mask (incl diag)
      }
    } else if (wave == 4) {
      // denom[s] = |n_s . q_s| clamped at 1 (uses OLD n_prev)
      const int s = lane;
      float d = 0.0f;
      for (int i2 = 0; i2 < HX; ++i2)
        d += Gs[s * HX + i2] * (np[i2] + Nr[s * HX + i2]) * qf[s * HX + i2];
      den[s] = fmaxf(fabsf(d), 1.0f);
    } else {
      // waves 5..7: async-copy next chunk's q,k,v,i,f slabs (5 x 16KB)
      // into LDS staging; each instruction moves 32 lanes x 16B = 512B.
      if (chn + 1 < NCH) {
        const int t0n = t0 + CS;
        const uint32_t lbase =
            (uint32_t)(uintptr_t)stg + (uint32_t)(lane * 16);
        for (int cc = wave - 5; cc < 5 * CS; cc += 3) {   // 160 x 512B blocks
          const int p  = cc >> 5;                         // gate plane 0..4
          const int w5 = (cc & 31) << 9;                  // offset in plane slab
          const uint64_t gaddr = (uint64_t)(uintptr_t)
              ((const char*)g +
               (((size_t)p * GSTRIDE + (size_t)(b * TX + t0n) * HX) << 2) +
               w5 + lane * 16);
          const uint32_t laddr = lbase + (uint32_t)(p * (CS * HX * 4) + w5);
          asm volatile("global_load_async_to_lds_b128 %0, %1, off"
                       :: "v"(laddr), "v"(gaddr) : "memory");
        }
      }
    }

    // H_inter(32x128) = Q C_prev^T : 16 tiles, 2 per wave
#pragma unroll
    for (int tt = 0; tt < 2; ++tt) {
      const int idx = wave * 2 + tt;
      const int tS = idx >> 3, tI = idx & 7;
      v8f c = {};
#pragma unroll
      for (int k0 = 0; k0 < HX; k0 += 32) {
        v16h a = load_afrag(Qh + (tS * 16 + l16) * HX, k0, hs);
        const float* crow = Cs + (tI * 16 + l16) * HX + k0 + hs * 16;
        v16h bb;
#pragma unroll
        for (int h = 0; h < 16; ++h) bb[h] = (_Float16)crow[h];
        c = wmma_f16(a, bb, c);
      }
#pragma unroll
      for (int r = 0; r < 8; ++r)
        Hi[(tS * 16 + r + hs * 8) * HX + tI * 16 + l16] = c[r];
    }
    __syncthreads();

    // ---------------- Phase D: outputs (intra term fused, f32 VALU) -----
#pragma unroll
    for (int r = 0; r < 16; ++r) {
      const int e = r * 256 + tid;
      const int s = e >> 7, i2 = e & 127;
      float acc = Hi[s * HX + i2];
      for (int u = 0; u < CS; ++u)
        acc = fmaf(Sc[s * CS + u], Uf[u * HX + i2], acc);
      const float ht = Gs[s * HX + i2] * acc;
      const size_t off = (size_t)(b * TX + t0 + s) * HX + i2;
      const float ov = g[5 * GSTRIDE + off];
      out[off] = ov * (ht / den[s]);
    }
    __syncthreads();

    // ---------------- Phase E: carry update C, n (WMMA K=32) ------------
#pragma unroll
    for (int tt = 0; tt < 8; ++tt) {
      const int idx = wave * 8 + tt;
      const int tI = idx >> 3, tJ = idx & 7;
      const int j = tJ * 16 + l16;
      v8f c;
#pragma unroll
      for (int r = 0; r < 8; ++r) {
        const int im = tI * 16 + r + hs * 8;
        c[r] = Ge[im] * Cs[im * HX + j];   // diag(G_end) * C_prev
      }
      v16h a = load_afrag(Vt + (tI * 16 + l16) * CS, 0, hs);
      v16h bb = *(const v16h*)(Kt + j * CS + hs * 16);
      c = wmma_f16(a, bb, c);              // += V'^T K
#pragma unroll
      for (int r = 0; r < 8; ++r) {
        const int im = tI * 16 + r + hs * 8;
        Cs[im * HX + j] = c[r];
      }
    }
    if (tid < HX)
      np[tid] = Ge[tid] * (np[tid] + Nr[(CS - 1) * HX + tid]);

    // issuing waves drain their async copies before anyone reads staging
    if (wave >= 5)
      asm volatile("s_wait_asynccnt 0x0" ::: "memory");
    __syncthreads();
  }
}

// ---------------------------------------------------------------------------
extern "C" void kernel_launch(void* const* d_in, const int* in_sizes, int n_in,
                              void* d_out, int out_size, void* d_ws, size_t ws_size,
                              hipStream_t stream) {
  (void)in_sizes; (void)n_in; (void)out_size; (void)ws_size;

  const float* x  = (const float*)d_in[0];
  const float* Wq = (const float*)d_in[1];
  const float* Wk = (const float*)d_in[2];
  const float* Wv = (const float*)d_in[3];
  const float* Wi = (const float*)d_in[4];
  const float* Wf = (const float*)d_in[5];
  const float* Wo = (const float*)d_in[6];
  const float* bq = (const float*)d_in[7];
  const float* bk = (const float*)d_in[8];
  const float* bv = (const float*)d_in[9];
  const float* bi = (const float*)d_in[10];
  const float* bf = (const float*)d_in[11];
  const float* bo = (const float*)d_in[12];

  char* ws = (char*)d_ws;
  _Float16* xh16 = (_Float16*)ws;
  _Float16* w16  = (_Float16*)(ws + (size_t)MTOT * DX * 2);
  float*    g    = (float*)(ws + (size_t)MTOT * DX * 2 + (size_t)6 * HX * DX * 2);
  float*    hbuf = g + 6 * GSTRIDE;

  const size_t lds_bytes = 185472 + 5 * CS * HX * 4;   // 267392

  const float* layin = x;
  for (int l = 0; l < 2; ++l) {
    cvt_f32_f16<<<(MTOT * DX + 255) / 256, 256, 0, stream>>>(layin, xh16, MTOT * DX);
    pack_w16<<<(6 * HX * DX + 255) / 256, 256, 0, stream>>>(Wq, Wk, Wv, Wi, Wf, Wo,
                                                            w16, l);
    proj_gemm<<<dim3(MTOT / 16, 6), 256, 0, stream>>>(xh16, w16,
                                                      bq, bk, bv, bi, bf, bo, g, l);
    float* o = (l == 1) ? (float*)d_out : hbuf;
    mlstm_chunk<<<BX, 256, lds_bytes, stream>>>(g, o);
    layin = hbuf;
  }
}
